// Encoder_ODE_RNN_24885040513716
// MI455X (gfx1250) — compile-verified, matrix operand order checked by reading, not run on hardware
//
#include <hip/hip_runtime.h>
#include <hip/hip_bf16.h>
#include <math.h>

typedef __attribute__((ext_vector_type(16))) __bf16 bf16x16;
typedef __attribute__((ext_vector_type(8)))  __bf16 bf16x8;
typedef __attribute__((ext_vector_type(4)))  __bf16 bf16x4;
typedef __attribute__((ext_vector_type(8)))  float  f32x8;

#define N_TRAJ 1024
#define N_TP   128
#define Y_DIM  64
#define H_DIM  256
#define R_DIM  128
#define MT     16          // trajectories per workgroup

// LDS row strides (bf16 elements, multiples of 8 => 16B aligned rows)
#define S_YB 336           // [y(256) | x(64) | pad]
#define S_GB 264           // gate tanh intermediates (256 + pad)
#define S_H  136           // ODE intermediates (128 + pad)

// Packed-weight offsets in workspace (bf16 elements). Tile = 32x16 = 512 bf16,
// stored [kt][nt][lane(32)][idx(16)]; lane fragment = 16 contiguous bf16 (32B).
#define OFF_WU1 0           // K=320 (KT=10), N=256 (NT=16): 81920
#define OFF_WR1 81920
#define OFF_WN1 163840
#define OFF_WU2 245760      // K=256 (8), N=256 (16): 65536
#define OFF_WR2 311296
#define OFF_WN2 376832
#define OFF_WO1 442368      // K=256 (8), N=128 (8): 32768   (N padded 100->128)
#define OFF_WO2 475136      // K=128 (4), N=128 (8): 16384   (K,N padded)
#define OFF_WO3 491520      // K=128 (4), N=256 (16): 32768  (K padded)
#define OFF_WFC 524288      // K=256 (8), N=128 (8): 32768
#define WS_ELEMS 557056     // ~1.06 MB bf16 total

// Bias table offsets inside LDS float array (ODE biases zero-padded to 128)
#define BB_U1 0
#define BB_U2 256
#define BB_R1 512
#define BB_R2 768
#define BB_N1 1024
#define BB_N2 1280
#define BB_O1 1536
#define BB_O2 1664
#define BB_O3 1792
#define BB_FC 2048
#define BB_TOT 2176

// ---------------------------------------------------------------------------
// Prep: fp32 [K,N] row-major -> packed bf16 WMMA B-fragment layout.
// ---------------------------------------------------------------------------
__global__ void pack_b_kernel(const float* __restrict__ src, __bf16* __restrict__ dst,
                              int Ksrc, int Nsrc, int KT, int NT) {
  int i = blockIdx.x * blockDim.x + threadIdx.x;
  int total = KT * NT * 512;
  if (i >= total) return;
  int idx  = i & 15;
  int lane = (i >> 4) & 31;
  int tile = i >> 9;
  int nt = tile % NT, kt = tile / NT;
  int hi = lane >> 4;
  int k_in = (idx & 7) + hi * 8 + ((idx & 8) ? 16 : 0);
  int k = kt * 32 + k_in;
  int n = nt * 16 + (lane & 15);
  float v = (k < Ksrc && n < Nsrc) ? src[k * Nsrc + n] : 0.0f;
  dst[i] = (__bf16)v;
}

// ---------------------------------------------------------------------------
// Fast activations (gfx1250 V_TANH_F32 confirmed in codegen).
// ---------------------------------------------------------------------------
__device__ __forceinline__ float fast_tanh(float x) {
#if __has_builtin(__builtin_amdgcn_tanhf)
  return __builtin_amdgcn_tanhf(x);
#elif __has_builtin(__builtin_amdgcn_tanh_f32)
  return __builtin_amdgcn_tanh_f32(x);
#else
  float e = __expf(2.0f * x);
  return 1.0f - 2.0f * __builtin_amdgcn_rcpf(e + 1.0f);
#endif
}
__device__ __forceinline__ float fast_sigmoid(float x) {
  return __builtin_amdgcn_rcpf(1.0f + __expf(-x));
}

// ---------------------------------------------------------------------------
// A-fragment preload: all KT fragments for this wave's 16-row A panel.
// ---------------------------------------------------------------------------
template <int KT>
__device__ __forceinline__ void load_afrags(const __bf16* A, int as, int lane,
                                            bf16x16* af) {
  const __bf16* arow = A + (lane & 15) * as + (lane >> 4) * 8;
#pragma unroll
  for (int kt = 0; kt < KT; ++kt) {
    bf16x8 lo = *(const bf16x8*)(arow + kt * 32);
    bf16x8 hh = *(const bf16x8*)(arow + kt * 32 + 16);
    af[kt] = __builtin_shufflevector(lo, hh, 0, 1, 2, 3, 4, 5, 6, 7,
                                     8, 9, 10, 11, 12, 13, 14, 15);
  }
}

// Single output tile; B double-buffered so load kt+1 overlaps wmma kt.
template <int KT>
__device__ __forceinline__ f32x8 gemm_b(const bf16x16* af,
                                        const __bf16* B,
                                        int NT, int nt, int lane) {
  f32x8 acc = {};
  const __bf16* bl = B + ((size_t)nt * 32 + (size_t)lane) * 16;
  bf16x16 bcur = *(const bf16x16*)(bl);
#pragma unroll
  for (int kt = 0; kt < KT; ++kt) {
    bf16x16 bnext = bcur;
    if (kt + 1 < KT)
      bnext = *(const bf16x16*)(bl + (size_t)(kt + 1) * NT * 512);
    acc = __builtin_amdgcn_wmma_f32_16x16x32_bf16(false, af[kt], false, bcur,
                                                  (short)0, acc, false, false);
    bcur = bnext;
  }
  return acc;
}

// Two output tiles (nt, nt+8) sharing A; interleaved chains, B double-buffered.
template <int KT>
__device__ __forceinline__ void gemm_b2(const bf16x16* af,
                                        const __bf16* B,
                                        int NT, int nt0, int lane,
                                        f32x8& acc0, f32x8& acc1) {
  acc0 = {};
  acc1 = {};
  const __bf16* bl0 = B + ((size_t)nt0 * 32 + (size_t)lane) * 16;
  const __bf16* bl1 = bl0 + 8 * 512;  // nt1 = nt0 + 8
  bf16x16 b0 = *(const bf16x16*)(bl0);
  bf16x16 b1 = *(const bf16x16*)(bl1);
#pragma unroll
  for (int kt = 0; kt < KT; ++kt) {
    bf16x16 n0 = b0, n1 = b1;
    if (kt + 1 < KT) {
      n0 = *(const bf16x16*)(bl0 + (size_t)(kt + 1) * NT * 512);
      n1 = *(const bf16x16*)(bl1 + (size_t)(kt + 1) * NT * 512);
    }
    acc0 = __builtin_amdgcn_wmma_f32_16x16x32_bf16(false, af[kt], false, b0,
                                                   (short)0, acc0, false, false);
    acc1 = __builtin_amdgcn_wmma_f32_16x16x32_bf16(false, af[kt], false, b1,
                                                   (short)0, acc1, false, false);
    b0 = n0;
    b1 = n1;
  }
}

// ---------------------------------------------------------------------------
// Persistent per-workgroup ODE-RNN scan: 16 trajectories, 128 steps, 8 waves.
// ---------------------------------------------------------------------------
__global__ __launch_bounds__(256, 1)
void odernn_kernel(const float* __restrict__ ts, const float* __restrict__ data,
                   const __bf16* __restrict__ ws,
                   const float* __restrict__ bu1, const float* __restrict__ bu2,
                   const float* __restrict__ br1, const float* __restrict__ br2,
                   const float* __restrict__ bn1, const float* __restrict__ bn2,
                   const float* __restrict__ bo1, const float* __restrict__ bo2,
                   const float* __restrict__ bo3, const float* __restrict__ bfc,
                   float* __restrict__ out) {
  __shared__ __bf16 YB[MT * S_YB];   // [ y_bf16(256) | x_bf16(64) | pad ]
  __shared__ float  Yf[MT * H_DIM];  // fp32 hidden state
  __shared__ __bf16 GB[MT * S_GB];   // tanh(u-gate L1) / tanh(n-gate L1)
  __shared__ __bf16 GB2[MT * S_GB];  // tanh(r-gate L1)
  __shared__ float  Uf[MT * H_DIM];  // update gate u
  __shared__ __bf16 H1[MT * S_H];    // ODE intermediates (padded to 128)
  __shared__ __bf16 H2[MT * S_H];
  __shared__ float  BIAS[BB_TOT];    // padded bias tables

  const int tid  = threadIdx.x;
  const int lane = tid & 31;
  const int w    = tid >> 5;       // wave id 0..7
  const int m    = lane & 15;
  const int hi   = lane >> 4;
  const int row0 = blockIdx.x * MT;

  // --- init state + bias tables -------------------------------------------
  for (int i = tid; i < MT * H_DIM; i += 256) Yf[i] = 0.0f;
  for (int i = tid; i < MT * S_YB; i += 256) YB[i] = (__bf16)0.0f;
  if (tid < 256) {
    int i = tid;
    BIAS[BB_U1 + i] = bu1[i]; BIAS[BB_U2 + i] = bu2[i];
    BIAS[BB_R1 + i] = br1[i]; BIAS[BB_R2 + i] = br2[i];
    BIAS[BB_N1 + i] = bn1[i]; BIAS[BB_N2 + i] = bn2[i];
    BIAS[BB_O3 + i] = bo3[i];
  }
  if (tid < 128) {
    int i = tid;
    BIAS[BB_O1 + i] = (i < 100) ? bo1[i] : 0.0f;
    BIAS[BB_O2 + i] = (i < 100) ? bo2[i] : 0.0f;
    BIAS[BB_FC + i] = bfc[i];
  }
  __syncthreads();

  for (int t = 0; t < N_TP; ++t) {
    // Launder a zero *offset* (not the pointer): defeats LICM of the
    // loop-invariant weight loads (which previously spilled to scratch)
    // while preserving pointer provenance so the loads stay on the
    // global path (no FLAT/DScnt coupling).
    size_t lz = 0;
    asm volatile("" : "+s"(lz));
    const __bf16* wsl = ws + lz;

    // dt schedule: first step -0.01, then ts[127-t] - ts[128-t] (run_backwards)
    float dt = (t == 0) ? -0.01f : (ts[127 - t] - ts[128 - t]);

    // load x_t = data[:, 127-t, :] into YB cols [256,320).
    // No barrier needed here: ODE-L1 only reads YB cols [0,256), and the
    // barrier after ODE-L1 orders these stores before u/r-L1 consumes x.
    {
      int r  = tid >> 4;            // 0..15
      int c4 = (tid & 15) << 2;     // 0,4,..,60
      const float* src =
          data + ((size_t)(row0 + r) * N_TP + (size_t)(127 - t)) * Y_DIM + c4;
      float4 v = *(const float4*)src;
      bf16x4 o;
      o.x = (__bf16)v.x; o.y = (__bf16)v.y; o.z = (__bf16)v.z; o.w = (__bf16)v.w;
      *(bf16x4*)&YB[r * S_YB + H_DIM + c4] = o;
    }

    // --- ODE L1: H1 = tanh(y @ Wo1 + bo1), K=256, N=128 (nt = w) -----------
    {
      bf16x16 af[8];
      load_afrags<8>(YB, S_YB, lane, af);
      f32x8 acc = gemm_b<8>(af, wsl + OFF_WO1, 8, w, lane);
      int col = w * 16 + m;
      float b = BIAS[BB_O1 + col];
#pragma unroll
      for (int v = 0; v < 8; ++v)
        H1[(hi * 8 + v) * S_H + col] = (__bf16)fast_tanh(acc[v] + b);
    }
    __syncthreads();

    // --- ODE L2: H2 = tanh(H1 @ Wo2 + bo2), K=128, N=128 -------------------
    {
      bf16x16 af[4];
      load_afrags<4>(H1, S_H, lane, af);
      f32x8 acc = gemm_b<4>(af, wsl + OFF_WO2, 8, w, lane);
      int col = w * 16 + m;
      float b = BIAS[BB_O2 + col];
#pragma unroll
      for (int v = 0; v < 8; ++v)
        H2[(hi * 8 + v) * S_H + col] = (__bf16)fast_tanh(acc[v] + b);
    }
    __syncthreads();

    // --- ODE L3 + Euler: y_ode = y + (H2 @ Wo3 + bo3)*dt, K=128, N=256 -----
    {
      bf16x16 af[4];
      load_afrags<4>(H2, S_H, lane, af);
      f32x8 acc[2];
      gemm_b2<4>(af, wsl + OFF_WO3, 16, w, lane, acc[0], acc[1]);
#pragma unroll
      for (int p = 0; p < 2; ++p) {
        int col = (w + p * 8) * 16 + m;
        float b = BIAS[BB_O3 + col];
#pragma unroll
        for (int v = 0; v < 8; ++v) {
          int row = hi * 8 + v;
          float yn = Yf[row * H_DIM + col] + (acc[p][v] + b) * dt;
          Yf[row * H_DIM + col] = yn;
          YB[row * S_YB + col] = (__bf16)yn;
        }
      }
    }
    __syncthreads();

    // --- u-L1 + r-L1 fused (shared A = [y_ode|x]), K=320, N=256 each -------
    {
      bf16x16 af[10];
      load_afrags<10>(YB, S_YB, lane, af);
      f32x8 au[2], ar[2];
      gemm_b2<10>(af, wsl + OFF_WU1, 16, w, lane, au[0], au[1]);
      gemm_b2<10>(af, wsl + OFF_WR1, 16, w, lane, ar[0], ar[1]);
#pragma unroll
      for (int p = 0; p < 2; ++p) {
        int col = (w + p * 8) * 16 + m;
        float bu = BIAS[BB_U1 + col];
        float br = BIAS[BB_R1 + col];
#pragma unroll
        for (int v = 0; v < 8; ++v) {
          int row = hi * 8 + v;
          GB[row * S_GB + col]  = (__bf16)fast_tanh(au[p][v] + bu);
          GB2[row * S_GB + col] = (__bf16)fast_tanh(ar[p][v] + br);
        }
      }
    }
    __syncthreads();

    // --- u-L2: u = sigmoid(GB @ Wu2 + bu2);  r-L2: YB = bf16(y_ode * r) ----
    {
      bf16x16 afu[8], afr[8];
      load_afrags<8>(GB, S_GB, lane, afu);
      load_afrags<8>(GB2, S_GB, lane, afr);
      f32x8 au[2], ar[2];
      gemm_b2<8>(afu, wsl + OFF_WU2, 16, w, lane, au[0], au[1]);
      gemm_b2<8>(afr, wsl + OFF_WR2, 16, w, lane, ar[0], ar[1]);
#pragma unroll
      for (int p = 0; p < 2; ++p) {
        int col = (w + p * 8) * 16 + m;
        float bu = BIAS[BB_U2 + col];
        float br = BIAS[BB_R2 + col];
#pragma unroll
        for (int v = 0; v < 8; ++v) {
          int row = hi * 8 + v;
          Uf[row * H_DIM + col] = fast_sigmoid(au[p][v] + bu);
          float r = fast_sigmoid(ar[p][v] + br);
          YB[row * S_YB + col] = (__bf16)(Yf[row * H_DIM + col] * r);
        }
      }
    }
    __syncthreads();

    // --- n-L1: GB = tanh([y*r|x] @ Wn1 + bn1), K=320, N=256 ----------------
    {
      bf16x16 af[10];
      load_afrags<10>(YB, S_YB, lane, af);
      f32x8 an[2];
      gemm_b2<10>(af, wsl + OFF_WN1, 16, w, lane, an[0], an[1]);
#pragma unroll
      for (int p = 0; p < 2; ++p) {
        int col = (w + p * 8) * 16 + m;
        float b = BIAS[BB_N1 + col];
#pragma unroll
        for (int v = 0; v < 8; ++v)
          GB[(hi * 8 + v) * S_GB + col] = (__bf16)fast_tanh(an[p][v] + b);
      }
    }
    __syncthreads();

    // --- n-L2 + GRU combine: y = (1-u)*n + u*y_ode -------------------------
    {
      bf16x16 af[8];
      load_afrags<8>(GB, S_GB, lane, af);
      f32x8 an[2];
      gemm_b2<8>(af, wsl + OFF_WN2, 16, w, lane, an[0], an[1]);
#pragma unroll
      for (int p = 0; p < 2; ++p) {
        int col = (w + p * 8) * 16 + m;
        float b = BIAS[BB_N2 + col];
#pragma unroll
        for (int v = 0; v < 8; ++v) {
          int row = hi * 8 + v;
          float n  = an[p][v] + b;
          float u  = Uf[row * H_DIM + col];
          float yo = Yf[row * H_DIM + col];
          float yn = (1.0f - u) * n + u * yo;
          Yf[row * H_DIM + col] = yn;
          YB[row * S_YB + col] = (__bf16)yn;
        }
      }
    }
    __syncthreads();
  }

  // --- output head: out = y @ Wfc + bfc, K=256, N=128 (nt = w) -------------
  {
    bf16x16 af[8];
    load_afrags<8>(YB, S_YB, lane, af);
    f32x8 acc = gemm_b<8>(af, ws + OFF_WFC, 8, w, lane);
    int col = w * 16 + m;
    float b = BIAS[BB_FC + col];
#pragma unroll
    for (int v = 0; v < 8; ++v)
      out[(size_t)(row0 + hi * 8 + v) * R_DIM + col] = acc[v] + b;
  }
}

extern "C" void kernel_launch(void* const* d_in, const int* in_sizes, int n_in,
                              void* d_out, int out_size, void* d_ws, size_t ws_size,
                              hipStream_t stream) {
  const float* data = (const float*)d_in[0];
  const float* ts   = (const float*)d_in[1];
  const float* Wu1 = (const float*)d_in[2];  const float* bu1 = (const float*)d_in[3];
  const float* Wu2 = (const float*)d_in[4];  const float* bu2 = (const float*)d_in[5];
  const float* Wr1 = (const float*)d_in[6];  const float* br1 = (const float*)d_in[7];
  const float* Wr2 = (const float*)d_in[8];  const float* br2 = (const float*)d_in[9];
  const float* Wn1 = (const float*)d_in[10]; const float* bn1 = (const float*)d_in[11];
  const float* Wn2 = (const float*)d_in[12]; const float* bn2 = (const float*)d_in[13];
  const float* Wo1 = (const float*)d_in[14]; const float* bo1 = (const float*)d_in[15];
  const float* Wo2 = (const float*)d_in[16]; const float* bo2 = (const float*)d_in[17];
  const float* Wo3 = (const float*)d_in[18]; const float* bo3 = (const float*)d_in[19];
  const float* Wfc = (const float*)d_in[20]; const float* bfc = (const float*)d_in[21];

  __bf16* ws = (__bf16*)d_ws;  // needs WS_ELEMS*2 ~= 1.07 MB of workspace

  auto pack = [&](const float* src, size_t off, int Ks, int Ns, int KT, int NT) {
    int total = KT * NT * 512;
    pack_b_kernel<<<(total + 255) / 256, 256, 0, stream>>>(src, ws + off, Ks, Ns, KT, NT);
  };
  pack(Wu1, OFF_WU1, 320, 256, 10, 16);
  pack(Wr1, OFF_WR1, 320, 256, 10, 16);
  pack(Wn1, OFF_WN1, 320, 256, 10, 16);
  pack(Wu2, OFF_WU2, 256, 256, 8, 16);
  pack(Wr2, OFF_WR2, 256, 256, 8, 16);
  pack(Wn2, OFF_WN2, 256, 256, 8, 16);
  pack(Wo1, OFF_WO1, 256, 100, 8, 8);
  pack(Wo2, OFF_WO2, 100, 100, 4, 8);
  pack(Wo3, OFF_WO3, 100, 256, 4, 16);
  pack(Wfc, OFF_WFC, 256, 128, 8, 8);

  odernn_kernel<<<N_TRAJ / MT, 256, 0, stream>>>(
      ts, data, ws, bu1, bu2, br1, br2, bn1, bn2, bo1, bo2, bo3, bfc,
      (float*)d_out);
}
